// TSLANet_52939766890823
// MI455X (gfx1250) — matrix operand order repaired and checked by reading, not compiled, not run on hardware
//
#include <hip/hip_runtime.h>
#include <math.h>

#define BB 64
#define LL 2048
#define MMCH 16
#define BMROWS (BB*MMCH)        // 1024
#define PATCHSZ 16
#define STRIDEP 8
#define NP 255
#define NPAD 256
#define EMB 128
#define FFREQ 128
#define DEPTHN 2
#define NBLK 8
#define BSZ 16
#define PRED 192
#define KOUT (EMB*NP)           // 32640
#define KSTEPS (KOUT/32)        // 1020
#define NROWS (BMROWS*FFREQ)    // 131072
#define NBINS 4096
#define LAMBDV 0.01f

typedef __attribute__((ext_vector_type(16))) _Float16 v16h;
typedef __attribute__((ext_vector_type(8)))  _Float16 v8h;
typedef __attribute__((ext_vector_type(8)))  float    v8f;
typedef __attribute__((ext_vector_type(4)))  float    v4f;
typedef __attribute__((ext_vector_type(4)))  int      v4i;

#define WMMA(a,b,c) __builtin_amdgcn_wmma_f32_16x16x32_f16(false,(a),false,(b),(short)0,(c),false,false)

// ---- CDNA5 async global->LDS DMA (ASYNCcnt path), guarded fallback ---------
#if defined(__has_builtin)
#if __has_builtin(__builtin_amdgcn_global_load_async_to_lds_b128) && \
    __has_builtin(__builtin_amdgcn_s_wait_asynccnt)
#define HAVE_ASYNC_LDS 1
#endif
#endif

__device__ __forceinline__ void async_copy16(void* lds, const void* gp) {
#ifdef HAVE_ASYNC_LDS
  __builtin_amdgcn_global_load_async_to_lds_b128((v4i*)gp, (v4i*)lds, 0, 0);
#else
  *(v8h*)lds = *(const v8h*)gp;
#endif
}
__device__ __forceinline__ void async_wait() {
#ifdef HAVE_ASYNC_LDS
  __builtin_amdgcn_s_wait_asynccnt(0);
#endif
}

__device__ __forceinline__ v16h cat16(v8h lo, v8h hi) {
  return __builtin_shufflevector(lo, hi, 0,1,2,3,4,5,6,7,8,9,10,11,12,13,14,15);
}

// A 16x32 f16 tile: per lane two contiguous 16B segments of row (lane&15):
//   halves [8*h2 .. +7] and [16+8*h2 .. +7]   -> 2x b128 loads
__device__ __forceinline__ v16h frag_a(const _Float16* A, int lda) {
  int l = threadIdx.x & 31; int r = l & 15; int h2 = l >> 4;
  const _Float16* p = A + r*lda + 8*h2;
  return cat16(*(const v8h*)p, *(const v8h*)(p + 16));
}
// B fragment pre-swizzled into fragment order [lane][16] -> 2x b128 loads
__device__ __forceinline__ v16h frag_bf(const _Float16* Bf) {
  int l = threadIdx.x & 31;
  const _Float16* p = Bf + l*16;
  return cat16(*(const v8h*)p, *(const v8h*)(p + 8));
}
// B fragment from a column-major (transposed) matrix: column fixed per lane,
// K contiguous -> 2x b128 loads.
__device__ __forceinline__ v16h frag_bt(const _Float16* colp) {
  int h2 = (threadIdx.x >> 4) & 1;
  const _Float16* p = colp + 8*h2;
  return cat16(*(const v8h*)p, *(const v8h*)(p + 16));
}

// ---- DFT / iDFT matrices (rfft & irfft as GEMMs, ortho norm) ---------------
__global__ void k_build_dft(_Float16* Fr, _Float16* Fi, _Float16* Gr, _Float16* Gi) {
  int i = blockIdx.x*blockDim.x + threadIdx.x;
  if (i >= FFREQ*NPAD) return;
  const float inv = 1.0f/sqrtf((float)NP);
  const float w0  = 6.28318530717958647692f/(float)NP;
  { // Fr/Fi : [FFREQ][NPAD]
    int f = i / NPAD, n = i % NPAD;
    float cr = 0.f, ci = 0.f;
    if (n < NP) { float ang = w0*(float)f*(float)n; cr = cosf(ang)*inv; ci = -sinf(ang)*inv; }
    Fr[i] = (_Float16)cr; Fi[i] = (_Float16)ci;
  }
  { // Gr/Gi : [NPAD][FFREQ]
    int n = i / FFREQ, f = i % FFREQ;
    float s = ((f == 0) ? 1.0f : 2.0f)*inv;
    float gr = 0.f, gi = 0.f;
    if (n < NP) { float ang = w0*(float)f*(float)n; gr = s*cosf(ang); gi = -s*sinf(ang); }
    Gr[i] = (_Float16)gr; Gi[i] = (_Float16)gi;
  }
}

// out_w swizzled to per-(kstep, coltile) fragment order
__global__ void k_swizzle_outw(const float* out_w, _Float16* owF) {
  int i = blockIdx.x*blockDim.x + threadIdx.x;
  if (i >= KSTEPS*12*512) return;
  int j  = i & 15;
  int l  = (i >> 4) & 31;
  int ct = (i >> 9) % 12;
  int ks = i / (512*12);
  int p = ct*16 + (l & 15);
  int k = ks*32 + ((j < 8) ? (8*(l >> 4) + j) : (16 + 8*(l >> 4) + (j - 8)));
  owF[i] = (_Float16)out_w[(size_t)p*KOUT + k];
}

// ---- instance norm statistics ----------------------------------------------
__global__ void k_meanstd(const float* x, float* meanp, float* stdp) {
  __shared__ float s1[256], s2[256];
  int bm = blockIdx.x; int b = bm >> 4, m = bm & 15;
  float sum = 0.f, sq = 0.f;
  for (int l = threadIdx.x; l < LL; l += 256) {
    float v = x[(b*LL + l)*MMCH + m];
    sum += v; sq += v*v;
  }
  s1[threadIdx.x] = sum; s2[threadIdx.x] = sq; __syncthreads();
  for (int o = 128; o > 0; o >>= 1) {
    if ((int)threadIdx.x < o) { s1[threadIdx.x] += s1[threadIdx.x+o]; s2[threadIdx.x] += s2[threadIdx.x+o]; }
    __syncthreads();
  }
  if (threadIdx.x == 0) {
    float mean = s1[0]/(float)LL;
    float var  = s2[0]/(float)LL - mean*mean;
    meanp[bm] = mean; stdp[bm] = sqrtf(var + 1e-5f);
  }
}

// ---- patch embed: h = xp @ in_w.T + in_b (K=16 padded to 32) ---------------
__global__ void k_patch_embed(const float* x, const float* in_w, const float* in_b,
                              const float* meanp, const float* stdp, float* h) {
  __shared__ _Float16 sA[16][32];
  __shared__ _Float16 sBf[8][32][16];     // fragment-order weights
  int bm = blockIdx.x, g = blockIdx.y;
  int b = bm >> 4, m = bm & 15;
  float mean = meanp[bm], istd = 1.0f/stdp[bm];
  int tid = threadIdx.x;
  {
    int row = tid >> 4, col = tid & 15;
    int t = g*16 + row;
    int l = t*STRIDEP + col; l = (l < LL) ? l : (LL - 1);   // clamp, branchless
    float val = (x[(b*LL + l)*MMCH + m] - mean)*istd;
    sA[row][col]      = (t < NP) ? (_Float16)val : (_Float16)0.f;
    sA[row][col + 16] = (_Float16)0.f;
  }
  for (int i = tid; i < 8*512; i += 256) {
    int wg = i >> 9; int rem = i & 511; int l = rem >> 4; int j = rem & 15;
    int k = (j < 8) ? (8*(l >> 4) + j) : (16 + 8*(l >> 4) + (j - 8));
    int e = wg*16 + (l & 15);
    sBf[wg][l][j] = (k < PATCHSZ) ? (_Float16)in_w[e*PATCHSZ + k] : (_Float16)0.f;
  }
  __syncthreads();
  int w = tid >> 5, lane = tid & 31, r = lane & 15, h2 = lane >> 4;
  v16h a = frag_a(&sA[0][0], 32);
  v16h bf = frag_bf(&sBf[w][0][0]);
  v8f acc = {};
  acc = WMMA(a, bf, acc);
  float bias = in_b[w*16 + r];                 // hoisted, loop-invariant
  float* hb = h + (size_t)bm*(NPAD*EMB) + (size_t)(g*16)*EMB + w*16 + r;
#pragma unroll
  for (int v = 0; v < 8; ++v) {
    int t = g*16 + v + 8*h2;
    hb[(v + 8*h2)*EMB] = (t < NP) ? (acc[v] + bias) : 0.0f;  // cndmask, no branch
  }
}

// ---- layernorm over EMB; emit TRANSPOSED f16: hfT[bm][c][n] ----------------
__global__ void k_layernorm_t(const float* h, const float* gamma, const float* beta,
                              _Float16* hfT, int d) {
  __shared__ float sH[16][EMB];
  __shared__ float sMean[16], sRstd[16];
  __shared__ float sPart[16][16];
  int g = blockIdx.x, bm = blockIdx.y;
  int tid = threadIdx.x;
  int row = tid >> 4, seg = tid & 15;
  const float* hb = h + (size_t)bm*(NPAD*EMB) + (size_t)(g*16)*EMB;
  v4f va = *(const v4f*)(hb + row*EMB + seg*8);
  v4f vb = *(const v4f*)(hb + row*EMB + seg*8 + 4);
  *(v4f*)&sH[row][seg*8]     = va;
  *(v4f*)&sH[row][seg*8 + 4] = vb;
  sPart[row][seg] = va[0]+va[1]+va[2]+va[3] + vb[0]+vb[1]+vb[2]+vb[3];
  __syncthreads();
  if (tid < 16) {
    float s = 0.f;
    for (int j = 0; j < 16; ++j) s += sPart[tid][j];
    sMean[tid] = s/(float)EMB;
  }
  __syncthreads();
  float mu = sMean[row];
  float vs = 0.f;
#pragma unroll
  for (int j = 0; j < 4; ++j) { float dA = va[j]-mu, dB = vb[j]-mu; vs += dA*dA + dB*dB; }
  sPart[row][seg] = vs;
  __syncthreads();
  if (tid < 16) {
    float s = 0.f;
    for (int j = 0; j < 16; ++j) s += sPart[tid][j];
    sRstd[tid] = rsqrtf(s/(float)EMB + 1e-5f);
  }
  __syncthreads();
  int e = tid >> 1, tseg = tid & 1;        // channel e, 8-token segment
  float ga = gamma[d*EMB + e], be = beta[d*EMB + e];
  v8h ov;
#pragma unroll
  for (int j = 0; j < 8; ++j) {
    int tr = tseg*8 + j;
    int t = g*16 + tr;
    float val = (sH[tr][e] - sMean[tr])*sRstd[tr]*ga + be;
    ov[j] = (t < NP) ? (_Float16)val : (_Float16)0.f;
  }
  *(v8h*)&hfT[((size_t)bm*EMB + e)*NPAD + g*16 + tseg*8] = ov;
}

// ---- rfft as GEMM: xf[bm,f,c] = (Fr + iFi)[f,:] @ ln[bm,:,c] ---------------
__global__ void k_rfft(const _Float16* hfT, const _Float16* Fr, const _Float16* Fi,
                       float* xr32T, float* xi32T, _Float16* xr16, _Float16* xi16) {
  __shared__ _Float16 sFr[16][NPAD];
  __shared__ _Float16 sFi[16][NPAD];
  int ft = blockIdx.x, bm = blockIdx.y;
  int tid = threadIdx.x;
  for (int i = tid; i < (16*NPAD)/8; i += 256) {      // async DMA staging
    int row = i >> 5, n8 = (i & 31)*8;
    async_copy16(&sFr[row][n8], Fr + (ft*16 + row)*NPAD + n8);
    async_copy16(&sFi[row][n8], Fi + (ft*16 + row)*NPAD + n8);
  }
  async_wait();
  __syncthreads();
  int w = tid >> 5, lane = tid & 31, r = lane & 15, h2 = lane >> 4;
  int c = w*16 + r;
  const _Float16* Bcol = hfT + ((size_t)bm*EMB + c)*NPAD;
  v8f accR = {}, accI = {};
#pragma unroll
  for (int kk = 0; kk < NPAD/32; ++kk) {
    v16h ar = frag_a(&sFr[0][kk*32], NPAD);
    v16h ai = frag_a(&sFi[0][kk*32], NPAD);
    v16h bf = frag_bt(Bcol + kk*32);
    accR = WMMA(ar, bf, accR);
    accI = WMMA(ai, bf, accI);
  }
  size_t tb = ((size_t)bm*EMB + c)*FFREQ + ft*16 + 8*h2;
  *(v4f*)(xr32T + tb)     = (v4f){accR[0], accR[1], accR[2], accR[3]};
  *(v4f*)(xr32T + tb + 4) = (v4f){accR[4], accR[5], accR[6], accR[7]};
  *(v4f*)(xi32T + tb)     = (v4f){accI[0], accI[1], accI[2], accI[3]};
  *(v4f*)(xi32T + tb + 4) = (v4f){accI[4], accI[5], accI[6], accI[7]};
#pragma unroll
  for (int v = 0; v < 8; ++v) {
    int f = ft*16 + v + 8*h2;
    int rowg = bm*FFREQ + f;
    xr16[(size_t)rowg*EMB + c] = (_Float16)accR[v];
    xi16[(size_t)rowg*EMB + c] = (_Float16)accI[v];
  }
}

__global__ void k_zero_hist(int* hist) {
  int i = blockIdx.x*blockDim.x + threadIdx.x;
  if (i < NBINS) hist[i] = 0;
}

// ---- spectral energy, per-bm median (bitonic), normalized energy + histogram
__global__ void k_energy(const float* xr32T, const float* xi32T, float* ne, int* hist) {
  __shared__ float s[FFREQ];
  int bm = blockIdx.x, f = threadIdx.x;
  const float* pr = xr32T + (size_t)bm*EMB*FFREQ + f;   // lane-coalesced over f
  const float* pi = xi32T + (size_t)bm*EMB*FFREQ + f;
  float sum = 0.f;
  for (int c = 0; c < EMB; ++c) {
    float a = pr[c*FFREQ], b = pi[c*FFREQ];
    sum += a*a + b*b;
  }
  s[f] = sum; __syncthreads();
  for (int k = 2; k <= FFREQ; k <<= 1)
    for (int j = k >> 1; j > 0; j >>= 1) {
      int ixj = f ^ j;
      if (ixj > f) {
        bool up = ((f & k) == 0);
        float a = s[f], b = s[ixj];
        if ((a > b) == up) { s[f] = b; s[ixj] = a; }
      }
      __syncthreads();
    }
  float med = 0.5f*(s[FFREQ/2 - 1] + s[FFREQ/2]);
  float nev = sum/(med + 1e-6f);
  ne[bm*FFREQ + f] = nev;
  int bin = (int)fminf(nev*64.0f, (float)(NBINS - 1));
  if (bin < 0) bin = 0;
  atomicAdd(&hist[bin], 1);
}

// ---- histogram CDF -> global quantile thresholds ---------------------------
__global__ void k_thresholds(const int* hist, const float* thr, int d, float* th) {
  if (threadIdx.x != 0 || blockIdx.x != 0) return;
  float pos_hi = thr[d]*(float)(NROWS - 1);
  float pos_lo = 0.5f*(float)(NROWS - 1);
  int cum = 0; float thi = (float)NBINS/64.0f, tlo = thi;
  bool fh = false, fl = false;
  for (int bi = 0; bi < NBINS; ++bi) {
    cum += hist[bi];
    float edge = (float)(bi + 1)/64.0f;
    if (!fh && (float)cum > pos_hi) { thi = edge; fh = true; }
    if (!fl && (float)cum > pos_lo) { tlo = edge; fl = true; }
    if (fh && fl) break;
  }
  th[0] = thi; th[1] = tlo;
}

// ---- block-diagonal complex 2-layer MLP + softshrink + spectral gating -----
__global__ void k_mlp_combine(const _Float16* xr16, const _Float16* xi16,
                              const float* xr32T, const float* xi32T,
                              const float* w1, const float* b1,
                              const float* w2, const float* b2,
                              const float* cw, const float* cwh,
                              const float* ne, const float* th,
                              _Float16* xwrT, _Float16* xwiT, int d) {
  __shared__ _Float16 sB1r[NBLK][32][16], sB1i[NBLK][32][16];   // [k][lane][j]
  __shared__ _Float16 sB2r[NBLK][32][16], sB2i[NBLK][32][16];
  __shared__ float sb1r[NBLK][16], sb1i[NBLK][16], sb2r[NBLK][16], sb2i[NBLK][16];
  __shared__ _Float16 so1[NBLK][16][32];
  int tid = threadIdx.x;
  const float* w1d = w1 + (size_t)d*2*NBLK*BSZ*BSZ;
  const float* w2d = w2 + (size_t)d*2*NBLK*BSZ*BSZ;
  const float* b1d = b1 + (size_t)d*2*NBLK*BSZ;
  const float* b2d = b2 + (size_t)d*2*NBLK*BSZ;
  for (int i = tid; i < NBLK*512; i += 256) {
    int k = i >> 9; int rem = i & 511; int l = rem >> 4; int j = rem & 15;
    int kk = (j < 8) ? (8*(l >> 4) + j) : (16 + 8*(l >> 4) + (j - 8));          // fragment-order swizzle
    int o  = l & 15;
    int ki = (kk < 16) ? kk : (kk - 16);
    float a1r = w1d[(0*NBLK + k)*256 + ki*16 + o];
    float a1i = w1d[(1*NBLK + k)*256 + ki*16 + o];
    float a2r = w2d[(0*NBLK + k)*256 + ki*16 + o];
    float a2i = w2d[(1*NBLK + k)*256 + ki*16 + o];
    if (kk < 16) { sB1r[k][l][j] = (_Float16)a1r;    sB1i[k][l][j] = (_Float16)a1i;
                   sB2r[k][l][j] = (_Float16)a2r;    sB2i[k][l][j] = (_Float16)a2i; }
    else         { sB1r[k][l][j] = (_Float16)(-a1i); sB1i[k][l][j] = (_Float16)a1r;
                   sB2r[k][l][j] = (_Float16)(-a2i); sB2i[k][l][j] = (_Float16)a2r; }
  }
  if (tid < NBLK*16) {
    int k = tid >> 4, o = tid & 15;
    sb1r[k][o] = b1d[(0*NBLK + k)*16 + o];
    sb1i[k][o] = b1d[(1*NBLK + k)*16 + o];
    sb2r[k][o] = b2d[(0*NBLK + k)*16 + o];
    sb2i[k][o] = b2d[(1*NBLK + k)*16 + o];
  }
  __syncthreads();
  int k = tid >> 5;                        // one wave per channel block
  int lane = tid & 31, r = lane & 15, h2 = lane >> 4;
  int r0 = blockIdx.x*16;
  int cb = k*BSZ;
  // layer1 A = [xr|xi]: concat boundary == segment boundary -> 2x b128
  const _Float16* pr = xr16 + (size_t)(r0 + r)*EMB + cb + 8*h2;
  const _Float16* pi = xi16 + (size_t)(r0 + r)*EMB + cb + 8*h2;
  v16h a1 = cat16(*(const v8h*)pr, *(const v8h*)pi);
  v8f o1r = {}, o1i = {};
  o1r = WMMA(a1, frag_bf(&sB1r[k][0][0]), o1r);
  o1i = WMMA(a1, frag_bf(&sB1i[k][0][0]), o1i);
#pragma unroll
  for (int v = 0; v < 8; ++v) {
    int row = v + 8*h2;
    float vr = o1r[v] + sb1r[k][r]; vr = vr > 0.f ? vr : 0.f;
    float vi = o1i[v] + sb1i[k][r]; vi = vi > 0.f ? vi : 0.f;
    so1[k][row][r]      = (_Float16)vr;   // same-wave LDS stays in order
    so1[k][row][16 + r] = (_Float16)vi;
  }
  v16h a2 = frag_a(&so1[k][0][0], 32);
  v8f o2r = {}, o2i = {};
  o2r = WMMA(a2, frag_bf(&sB2r[k][0][0]), o2r);
  o2i = WMMA(a2, frag_bf(&sB2i[k][0][0]), o2i);
  // combine, all transposed-layout vector accesses
  int bmM = r0 >> 7;
  int fb  = (r0 & 127) + 8*h2;
  int c   = cb + r;
  size_t tb = ((size_t)bmM*EMB + c)*FFREQ + fb;
  v4f xrA = *(const v4f*)(xr32T + tb), xrB = *(const v4f*)(xr32T + tb + 4);
  v4f xiA = *(const v4f*)(xi32T + tb), xiB = *(const v4f*)(xi32T + tb + 4);
  v4f neA = *(const v4f*)(ne + r0 + 8*h2), neB = *(const v4f*)(ne + r0 + 8*h2 + 4);
  float cwr = cw[(d*EMB + c)*2 + 0],  cwi = cw[(d*EMB + c)*2 + 1];
  float whr = cwh[(d*EMB + c)*2 + 0], whi = cwh[(d*EMB + c)*2 + 1];
  float th_hi = th[0], th_lo = th[1];
  v8h owr, owi;
#pragma unroll
  for (int v = 0; v < 8; ++v) {
    float lr = o2r[v] + sb2r[k][r];
    float li = o2i[v] + sb2i[k][r];
    lr = (lr > LAMBDV) ? (lr - LAMBDV) : ((lr < -LAMBDV) ? (lr + LAMBDV) : 0.f);
    li = (li > LAMBDV) ? (li - LAMBDV) : ((li < -LAMBDV) ? (li + LAMBDV) : 0.f);
    float xr = (v < 4) ? xrA[v] : xrB[v-4];
    float xi = (v < 4) ? xiA[v] : xiB[v-4];
    float nev = (v < 4) ? neA[v] : neB[v-4];
    float fac = (nev > th_hi) ? ((nev <= th_lo) ? 2.f : 1.f) : 0.f;
    float swr = xr*cwr - xi*cwi + fac*(xr*whr - xi*whi);   // S = xf*(cw+fac*wh)
    float swi = xr*cwi + xi*cwr + fac*(xr*whi + xi*whr);
    float xlr = lr*xr - li*xi;                             // x_lin = o2*xf
    float xli = lr*xi + li*xr;
    owr[v] = (_Float16)(swr*xlr - swi*xli);                // xw = S*x_lin
    owi[v] = (_Float16)(swr*xli + swi*xlr);
  }
  *(v8h*)(xwrT + tb) = owr;   // single 16B store each
  *(v8h*)(xwiT + tb) = owi;
}

// ---- irfft as GEMM + residual add into h (unguarded RMW; padded row is
// write-only garbage that every consumer masks) ------------------------------
__global__ void k_irfft_res(const _Float16* xwrT, const _Float16* xwiT,
                            const _Float16* Gr, const _Float16* Gi, float* h) {
  __shared__ _Float16 sGr[16][FFREQ], sGi[16][FFREQ];
  int g = blockIdx.x, bm = blockIdx.y;
  int tid = threadIdx.x;
  for (int i = tid; i < (16*FFREQ)/8; i += 256) {     // async DMA staging
    int row = i >> 4, f8 = (i & 15)*8;
    async_copy16(&sGr[row][f8], Gr + (g*16 + row)*FFREQ + f8);
    async_copy16(&sGi[row][f8], Gi + (g*16 + row)*FFREQ + f8);
  }
  async_wait();
  __syncthreads();
  int w = tid >> 5, lane = tid & 31, r = lane & 15, h2 = lane >> 4;
  int c = w*16 + r;
  const _Float16* BrCol = xwrT + ((size_t)bm*EMB + c)*FFREQ;
  const _Float16* BiCol = xwiT + ((size_t)bm*EMB + c)*FFREQ;
  v8f acc = {};
#pragma unroll
  for (int kk = 0; kk < FFREQ/32; ++kk) {
    v16h ar = frag_a(&sGr[0][kk*32], FFREQ);
    v16h ai = frag_a(&sGi[0][kk*32], FFREQ);
    v16h br = frag_bt(BrCol + kk*32);
    v16h bi = frag_bt(BiCol + kk*32);
    acc = WMMA(ar, br, acc);
    acc = WMMA(ai, bi, acc);
  }
  float* hb = h + (size_t)bm*(NPAD*EMB) + (size_t)(g*16)*EMB + c;
  float hv[8];
#pragma unroll
  for (int v = 0; v < 8; ++v) hv[v] = hb[(v + 8*h2)*EMB];
#pragma unroll
  for (int v = 0; v < 8; ++v) hb[(v + 8*h2)*EMB] = hv[v] + acc[v];
}

__global__ void k_cast_h(const float* h, _Float16* hf) {
  int i = blockIdx.x*blockDim.x + threadIdx.x;
  if (i >= BMROWS*NPAD*EMB) return;
  int t = (i >> 7) & 255;
  float hv = h[i];                                  // unconditional load
  hf[i] = (t < NP) ? (_Float16)hv : (_Float16)0.f;  // value select only
}

// ---- output head: out_acc = h.flat @ out_w.T + out_b (K = 32640) -----------
__global__ void k_head(const _Float16* hf, const _Float16* owF, const float* out_b,
                       float* out_acc) {
  int tid = threadIdx.x;                 // 128 threads = 4 waves
  int w = tid >> 5, lane = tid & 31, r = lane & 15, h2 = lane >> 4;
  int ct = blockIdx.x*4 + w;             // 0..11 col tiles
  int rt = blockIdx.y;                   // 0..63 row tiles
  const _Float16* Abase = hf + (size_t)(rt*16 + r)*(NPAD*EMB) + 8*h2;
  const _Float16* Bbase = owF + ((size_t)ct*32 + lane)*16;
  v8f acc = {};
  for (int ks = 0; ks < KSTEPS; ++ks) {
    const _Float16* pa = Abase + ks*32;
    v16h a = cat16(*(const v8h*)pa, *(const v8h*)(pa + 16));
    const _Float16* pb = Bbase + (size_t)ks*(12*512);
    v16h bf = cat16(*(const v8h*)pb, *(const v8h*)(pb + 8));
    __builtin_prefetch(pb + 12*512, 0, 1);
    acc = WMMA(a, bf, acc);
  }
#pragma unroll
  for (int v = 0; v < 8; ++v) {
    int bm = rt*16 + v + 8*h2;
    int p = ct*16 + r;
    out_acc[bm*PRED + p] = acc[v] + out_b[p];
  }
}

__global__ void k_final(const float* out_acc, const float* meanp, const float* stdp,
                        float* out) {
  int i = blockIdx.x*blockDim.x + threadIdx.x;
  if (i >= BB*PRED*MMCH) return;
  int m = i % MMCH;
  int p = (i / MMCH) % PRED;
  int b = i / (MMCH*PRED);
  int bm = b*MMCH + m;
  out[i] = out_acc[bm*PRED + p]*stdp[bm] + meanp[bm];
}

extern "C" void kernel_launch(void* const* d_in, const int* in_sizes, int n_in,
                              void* d_out, int out_size, void* d_ws, size_t ws_size,
                              hipStream_t stream) {
  (void)in_sizes; (void)n_in; (void)out_size; (void)ws_size;
  const float* x     = (const float*)d_in[0];
  const float* in_w  = (const float*)d_in[1];
  const float* in_b  = (const float*)d_in[2];
  const float* gamma = (const float*)d_in[3];
  const float* beta  = (const float*)d_in[4];
  const float* cw    = (const float*)d_in[5];
  const float* cwh   = (const float*)d_in[6];
  const float* w1    = (const float*)d_in[7];
  const float* b1    = (const float*)d_in[8];
  const float* w2    = (const float*)d_in[9];
  const float* b2    = (const float*)d_in[10];
  const float* thr   = (const float*)d_in[11];
  const float* out_w = (const float*)d_in[12];
  const float* out_b = (const float*)d_in[13];
  float* out = (float*)d_out;

  char* ws = (char*)d_ws;
  size_t off = 0;
  auto alloc = [&](size_t bytes) -> void* {
    void* p = ws + off; off += (bytes + 255) & ~(size_t)255; return p;
  };
  float*    meanp  = (float*)   alloc((size_t)BMROWS*4);
  float*    stdp   = (float*)   alloc((size_t)BMROWS*4);
  float*    th     = (float*)   alloc(2*4);
  int*      hist   = (int*)     alloc((size_t)NBINS*4);
  _Float16* Fr     = (_Float16*)alloc((size_t)FFREQ*NPAD*2);
  _Float16* Fi     = (_Float16*)alloc((size_t)FFREQ*NPAD*2);
  _Float16* Gr     = (_Float16*)alloc((size_t)NPAD*FFREQ*2);
  _Float16* Gi     = (_Float16*)alloc((size_t)NPAD*FFREQ*2);
  _Float16* owF    = (_Float16*)alloc((size_t)KOUT*PRED*2);
  float*    h      = (float*)   alloc((size_t)BMROWS*NPAD*EMB*4);
  _Float16* hf     = (_Float16*)alloc((size_t)BMROWS*NPAD*EMB*2);
  _Float16* hfT    = (_Float16*)alloc((size_t)BMROWS*EMB*NPAD*2);
  float*    xr32T  = (float*)   alloc((size_t)BMROWS*EMB*FFREQ*4);
  float*    xi32T  = (float*)   alloc((size_t)BMROWS*EMB*FFREQ*4);
  _Float16* xr16   = (_Float16*)alloc((size_t)NROWS*EMB*2);
  _Float16* xi16   = (_Float16*)alloc((size_t)NROWS*EMB*2);
  _Float16* xwrT   = (_Float16*)alloc((size_t)BMROWS*EMB*FFREQ*2);
  _Float16* xwiT   = (_Float16*)alloc((size_t)BMROWS*EMB*FFREQ*2);
  float*    ne     = (float*)   alloc((size_t)NROWS*4);
  float*    outacc = (float*)   alloc((size_t)BMROWS*PRED*4);

  k_build_dft<<<(FFREQ*NPAD + 255)/256, 256, 0, stream>>>(Fr, Fi, Gr, Gi);
  k_swizzle_outw<<<(KOUT*PRED + 255)/256, 256, 0, stream>>>(out_w, owF);
  k_meanstd<<<BMROWS, 256, 0, stream>>>(x, meanp, stdp);
  k_patch_embed<<<dim3(BMROWS, NPAD/16), 256, 0, stream>>>(x, in_w, in_b, meanp, stdp, h);
  for (int d = 0; d < DEPTHN; ++d) {
    k_layernorm_t<<<dim3(NPAD/16, BMROWS), 256, 0, stream>>>(h, gamma, beta, hfT, d);
    k_rfft<<<dim3(FFREQ/16, BMROWS), 256, 0, stream>>>(hfT, Fr, Fi, xr32T, xi32T, xr16, xi16);
    k_zero_hist<<<(NBINS + 255)/256, 256, 0, stream>>>(hist);
    k_energy<<<BMROWS, FFREQ, 0, stream>>>(xr32T, xi32T, ne, hist);
    k_thresholds<<<1, 1, 0, stream>>>(hist, thr, d, th);
    k_mlp_combine<<<NROWS/16, 256, 0, stream>>>(xr16, xi16, xr32T, xi32T,
                                                w1, b1, w2, b2, cw, cwh, ne, th,
                                                xwrT, xwiT, d);
    k_irfft_res<<<dim3(NPAD/16, BMROWS), 256, 0, stream>>>(xwrT, xwiT, Gr, Gi, h);
  }
  k_cast_h<<<(BMROWS*NPAD*EMB + 255)/256, 256, 0, stream>>>(h, hf);
  k_head<<<dim3(3, BMROWS/16), 128, 0, stream>>>(hf, owF, out_b, outacc);
  k_final<<<(BB*PRED*MMCH + 255)/256, 256, 0, stream>>>(outacc, meanp, stdp, out);
}
